// AfmoeAttention_79087527789057
// MI455X (gfx1250) — matrix operand order repaired and checked by reading, not compiled, not run on hardware
//
#include <hip/hip_runtime.h>
#include <hip/hip_bf16.h>

typedef __bf16 bf16;
typedef __attribute__((ext_vector_type(16))) __bf16 v16bf;
typedef __attribute__((ext_vector_type(8)))  __bf16 v8bf;
typedef __attribute__((ext_vector_type(8)))  float  v8f;

// Problem constants (from reference): B=2, S=2048, HID=2048, NH=32, NKV=4, D=64
#define CB   2
#define CS   2048
#define CHID 2048
#define CNH  32
#define CNKV 4
#define CD   64
#define CM   (CB * CS)          // 4096 rows of tokens
#define CEPS 1e-6f

// ---------------------------------------------------------------------------
// Fragment loaders matching CDNA5 WMMA VGPR layouts (CDNA5 ISA):
//  A (16x32 bf16): lane L<16 holds row M=L, K={kh*8..kh*8+7, 16+kh*8..23+kh*8}
//  B (32x16 bf16): lane L holds col N=L%16, K={kh*8..kh*8+7, 16+kh*8..23+kh*8}
// ---------------------------------------------------------------------------
__device__ __forceinline__ v16bf load_a(const bf16* p, int kh) {
  v8bf c0 = *(const v8bf*)(p + kh * 8);
  v8bf c1 = *(const v8bf*)(p + 16 + kh * 8);
  v16bf f;
#pragma unroll
  for (int i = 0; i < 8; ++i) { f[i] = c0[i]; f[i + 8] = c1[i]; }
  return f;
}

__device__ __forceinline__ v16bf load_b(const bf16* p, int kh) {
  v8bf c0 = *(const v8bf*)(p + kh * 8);
  v8bf c1 = *(const v8bf*)(p + 16 + kh * 8);
  v16bf f;
#pragma unroll
  for (int i = 0; i < 8; ++i) { f[i] = c0[i]; f[i + 8] = c1[i]; }
  return f;
}

__device__ __forceinline__ v8f wmma_bf(v16bf a, v16bf b, v8f c) {
  // v_wmma_f32_16x16x32_bf16
  return __builtin_amdgcn_wmma_f32_16x16x32_bf16(false, a, false, b,
                                                 (short)0, c, false, false);
}

// ---------------------------------------------------------------------------
// fp32 -> bf16 conversion (vectorized x4)
// ---------------------------------------------------------------------------
__global__ void k_cvt_bf16(const float* __restrict__ in, bf16* __restrict__ out, int n) {
  int i = (blockIdx.x * blockDim.x + threadIdx.x) * 4;
  if (i + 3 < n) {
    float4 v = *(const float4*)(in + i);
    out[i + 0] = (bf16)v.x;
    out[i + 1] = (bf16)v.y;
    out[i + 2] = (bf16)v.z;
    out[i + 3] = (bf16)v.w;
  } else {
    for (int j = i; j < n; ++j) out[j] = (bf16)in[j];
  }
}

// ---------------------------------------------------------------------------
// GEMM: C[M,N] = A[M,K] * W[N,K]^T  (bf16 in, fp32 out)
// Block 256 (8 waves). Wave tile 32(M) x 64(N). Block tile 128x128.
// Working set is L2-resident (192MB) -> stream global->register, no LDS stage.
// ---------------------------------------------------------------------------
__global__ __launch_bounds__(256) void k_gemm_bf16(
    const bf16* __restrict__ A, const bf16* __restrict__ W,
    float* __restrict__ C, int M, int N, int K) {
  const int lane = threadIdx.x & 31;
  const int w    = threadIdx.x >> 5;
  const int r16  = lane & 15;
  const int kh   = lane >> 4;

  const int mBase = blockIdx.y * 128 + (w & 3) * 32;
  const int nBase = blockIdx.x * 128 + (w >> 2) * 64;

  const bf16* aRow0 = A + (size_t)(mBase + r16) * K;
  const bf16* aRow1 = A + (size_t)(mBase + 16 + r16) * K;
  const bf16* bRow0 = W + (size_t)(nBase + 0 * 16 + r16) * K;
  const bf16* bRow1 = W + (size_t)(nBase + 1 * 16 + r16) * K;
  const bf16* bRow2 = W + (size_t)(nBase + 2 * 16 + r16) * K;
  const bf16* bRow3 = W + (size_t)(nBase + 3 * 16 + r16) * K;

  v8f acc[2][4] = {};

  for (int k0 = 0; k0 < K; k0 += 32) {
    // prefetch next k-step into the hierarchy
    if (k0 + 32 < K) {
      __builtin_prefetch(aRow0 + k0 + 32, 0, 3);
      __builtin_prefetch(bRow0 + k0 + 32, 0, 3);
    }
    v16bf a0 = load_a(aRow0 + k0, kh);
    v16bf a1 = load_a(aRow1 + k0, kh);
    v16bf b0 = load_b(bRow0 + k0, kh);
    v16bf b1 = load_b(bRow1 + k0, kh);
    v16bf b2 = load_b(bRow2 + k0, kh);
    v16bf b3 = load_b(bRow3 + k0, kh);
    acc[0][0] = wmma_bf(a0, b0, acc[0][0]);
    acc[1][0] = wmma_bf(a1, b0, acc[1][0]);
    acc[0][1] = wmma_bf(a0, b1, acc[0][1]);
    acc[1][1] = wmma_bf(a1, b1, acc[1][1]);
    acc[0][2] = wmma_bf(a0, b2, acc[0][2]);
    acc[1][2] = wmma_bf(a1, b2, acc[1][2]);
    acc[0][3] = wmma_bf(a0, b3, acc[0][3]);
    acc[1][3] = wmma_bf(a1, b3, acc[1][3]);
  }

  // C/D layout: element j of lane L -> row = j + 8*(L/16), col = L%16
#pragma unroll
  for (int i = 0; i < 2; ++i)
#pragma unroll
    for (int t = 0; t < 4; ++t)
#pragma unroll
      for (int j = 0; j < 8; ++j)
        C[(size_t)(mBase + i * 16 + 8 * kh + j) * N + nBase + t * 16 + r16] =
            acc[i][t][j];
}

// ---------------------------------------------------------------------------
// Per-head RMSNorm + RoPE; writes head-major bf16 tensors for attention.
//  - Q: scaled by D^-0.5 = 0.125 (exact power of two, safe in bf16)
//  - Q out: [b][h][s][d], K out: [b][kv][s][d], V out transposed: [b][kv][d][s]
// Thread = one (token, head-slot); slot 0..31 = Q heads, 32..35 = K, 36..39 = V
// ---------------------------------------------------------------------------
__global__ void k_norm_rope(const float* __restrict__ qf, const float* __restrict__ kf,
                            const float* __restrict__ vf,
                            const float* __restrict__ cosb, const float* __restrict__ sinb,
                            const float* __restrict__ qg, const float* __restrict__ kg,
                            bf16* __restrict__ qT, bf16* __restrict__ kT,
                            bf16* __restrict__ vT) {
  int tid  = blockIdx.x * blockDim.x + threadIdx.x;
  if (tid >= CM * (CNH + 2 * CNKV)) return;
  int tok  = tid / (CNH + 2 * CNKV);
  int slot = tid % (CNH + 2 * CNKV);
  int b = tok / CS, s = tok % CS;
  const float* cs = cosb + (size_t)tok * CD;
  const float* sn = sinb + (size_t)tok * CD;

  if (slot < CNH) {
    int h = slot;
    const float* x = qf + (size_t)tok * (CNH * CD) + (size_t)h * CD;
    float v[CD]; float ss = 0.f;
#pragma unroll
    for (int d = 0; d < CD; ++d) { v[d] = x[d]; ss += v[d] * v[d]; }
    float inv = rsqrtf(ss * (1.0f / CD) + CEPS) * 0.125f;  // fold score scale
    bf16* o = qT + ((size_t)(b * CNH + h) * CS + s) * CD;
#pragma unroll
    for (int d = 0; d < CD / 2; ++d) {
      float xn = v[d] * inv * qg[d];
      float xr = v[d + CD / 2] * inv * qg[d + CD / 2];
      o[d]           = (bf16)(xn * cs[d] - xr * sn[d]);
      o[d + CD / 2]  = (bf16)(xr * cs[d + CD / 2] + xn * sn[d + CD / 2]);
    }
  } else if (slot < CNH + CNKV) {
    int kv = slot - CNH;
    const float* x = kf + (size_t)tok * (CNKV * CD) + (size_t)kv * CD;
    float v[CD]; float ss = 0.f;
#pragma unroll
    for (int d = 0; d < CD; ++d) { v[d] = x[d]; ss += v[d] * v[d]; }
    float inv = rsqrtf(ss * (1.0f / CD) + CEPS);
    bf16* o = kT + ((size_t)(b * CNKV + kv) * CS + s) * CD;
#pragma unroll
    for (int d = 0; d < CD / 2; ++d) {
      float xn = v[d] * inv * kg[d];
      float xr = v[d + CD / 2] * inv * kg[d + CD / 2];
      o[d]          = (bf16)(xn * cs[d] - xr * sn[d]);
      o[d + CD / 2] = (bf16)(xr * cs[d + CD / 2] + xn * sn[d + CD / 2]);
    }
  } else {
    int kv = slot - CNH - CNKV;
    const float* x = vf + (size_t)tok * (CNKV * CD) + (size_t)kv * CD;
    bf16* o = vT + (size_t)(b * CNKV + kv) * CD * CS + s;  // [b][kv][d][s]
#pragma unroll
    for (int d = 0; d < CD; ++d) o[(size_t)d * CS] = (bf16)x[d];
  }
}

// ---------------------------------------------------------------------------
// Flash attention (non-causal), one (b,h) per blockIdx.y, 128 q-rows / block.
// Wave: 16 q-rows x 64 dims of output; streams 64-key blocks.
// Scores: 8 WMMA; online softmax via 16-lane shfl_xor (matches C-layout row
// split across wave32 halves); P spilled to wave-private LDS to transpose the
// C-layout into an A-fragment layout for P@V (8 more WMMA).
// ---------------------------------------------------------------------------
__global__ __launch_bounds__(256) void k_flash(
    const bf16* __restrict__ qT, const bf16* __restrict__ kT,
    const bf16* __restrict__ vT, float* __restrict__ att) {
  __shared__ bf16 pShared[8 * 16 * 64];  // 16 KB, 2 KB per wave

  const int lane = threadIdx.x & 31;
  const int w    = threadIdx.x >> 5;
  const int r16  = lane & 15;
  const int kh   = lane >> 4;

  const int bh = blockIdx.y;            // b*NH + h
  const int b  = bh / CNH;
  const int h  = bh % CNH;
  const int kv = h / (CNH / CNKV);
  const int qRow0 = blockIdx.x * 128 + w * 16;

  const bf16* qp = qT + ((size_t)bh * CS + qRow0 + r16) * CD;
  v16bf aq0 = load_a(qp, kh);        // d = 0..31
  v16bf aq1 = load_a(qp + 32, kh);   // d = 32..63

  const bf16* kbase = kT + (size_t)(b * CNKV + kv) * CS * CD;  // [key][d]
  const bf16* vbase = vT + (size_t)(b * CNKV + kv) * CD * CS;  // [d][key]

  float mrow[8], lrow[8];
  v8f acc[4] = {};
#pragma unroll
  for (int j = 0; j < 8; ++j) { mrow[j] = -3.0e38f; lrow[j] = 0.f; }

  bf16* pw = pShared + w * (16 * 64);

  for (int kb = 0; kb < CS; kb += 64) {
    // ---- scores: 16 q-rows x 64 keys ----
    v8f c[4];
#pragma unroll
    for (int t = 0; t < 4; ++t) {
      const bf16* kr = kbase + (size_t)(kb + t * 16 + r16) * CD;
      v16bf bk0 = load_b(kr, kh);
      v16bf bk1 = load_b(kr + 32, kh);
      v8f z = {};
      z = wmma_bf(aq0, bk0, z);
      z = wmma_bf(aq1, bk1, z);
      c[t] = z;
    }

    // ---- online softmax (rows split: lanes 0-15 = rows 0-7, 16-31 = 8-15) --
#pragma unroll
    for (int j = 0; j < 8; ++j) {
      float t0 = fmaxf(fmaxf(c[0][j], c[1][j]), fmaxf(c[2][j], c[3][j]));
      t0 = fmaxf(t0, __shfl_xor(t0, 1, 32));
      t0 = fmaxf(t0, __shfl_xor(t0, 2, 32));
      t0 = fmaxf(t0, __shfl_xor(t0, 4, 32));
      t0 = fmaxf(t0, __shfl_xor(t0, 8, 32));
      float mn    = fmaxf(mrow[j], t0);
      float alpha = __expf(mrow[j] - mn);
      float s = 0.f;
#pragma unroll
      for (int t = 0; t < 4; ++t) {
        float p = __expf(c[t][j] - mn);
        c[t][j] = p;
        s += p;
      }
      s += __shfl_xor(s, 1, 32);
      s += __shfl_xor(s, 2, 32);
      s += __shfl_xor(s, 4, 32);
      s += __shfl_xor(s, 8, 32);
      lrow[j] = lrow[j] * alpha + s;
#pragma unroll
      for (int t = 0; t < 4; ++t) acc[t][j] *= alpha;
      mrow[j] = mn;
    }

    // ---- transpose P via wave-private LDS into A-fragment layout ----
#pragma unroll
    for (int t = 0; t < 4; ++t)
#pragma unroll
      for (int j = 0; j < 8; ++j)
        pw[(j + 8 * kh) * 64 + t * 16 + r16] = (bf16)c[t][j];
    asm volatile("s_wait_dscnt 0" ::: "memory");
    v16bf p0 = load_a(pw + r16 * 64, kh);       // keys 0..31
    v16bf p1 = load_a(pw + r16 * 64 + 32, kh);  // keys 32..63

    // ---- acc += P @ V (V transposed so B-fragments are contiguous) ----
#pragma unroll
    for (int t = 0; t < 4; ++t) {
      const bf16* vr = vbase + (size_t)(t * 16 + r16) * CS + kb;
      v16bf bv0 = load_b(vr, kh);
      v16bf bv1 = load_b(vr + 32, kh);
      acc[t] = wmma_bf(p0, bv0, acc[t]);
      acc[t] = wmma_bf(p1, bv1, acc[t]);
    }
  }

  // ---- normalize and store fp32, layout [b][s][h*64+d] ----
#pragma unroll
  for (int j = 0; j < 8; ++j) {
    float invl = 1.0f / lrow[j];
    size_t row = (size_t)b * CS + qRow0 + j + 8 * kh;
    float* op = att + row * (CNH * CD) + (size_t)h * CD;
#pragma unroll
    for (int t = 0; t < 4; ++t) op[t * 16 + r16] = acc[t][j] * invl;
  }
}

// ---------------------------------------------------------------------------
// gated[i] = bf16( att[i] * sigmoid(gate[i]) )
// ---------------------------------------------------------------------------
__global__ void k_gate(const float* __restrict__ att, const float* __restrict__ gate,
                       bf16* __restrict__ out, int n) {
  int i = blockIdx.x * blockDim.x + threadIdx.x;
  if (i < n) {
    float g  = gate[i];
    float sg = 1.0f / (1.0f + __expf(-g));
    out[i] = (bf16)(att[i] * sg);
  }
}

// ---------------------------------------------------------------------------
extern "C" void kernel_launch(void* const* d_in, const int* in_sizes, int n_in,
                              void* d_out, int out_size, void* d_ws, size_t ws_size,
                              hipStream_t stream) {
  (void)in_sizes; (void)n_in; (void)out_size; (void)ws_size;
  const float* hidden = (const float*)d_in[0];
  const float* cosb   = (const float*)d_in[1];
  const float* sinb   = (const float*)d_in[2];
  const float* Wq     = (const float*)d_in[3];
  const float* Wk     = (const float*)d_in[4];
  const float* Wv     = (const float*)d_in[5];
  const float* Wg     = (const float*)d_in[6];
  const float* Wo     = (const float*)d_in[7];
  const float* qg     = (const float*)d_in[8];
  const float* kg     = (const float*)d_in[9];
  float* out = (float*)d_out;

  char* p = (char*)d_ws;
  auto nxt = [&](size_t bytes) {
    char* r = p;
    p += (bytes + 255) & ~(size_t)255;
    return r;
  };

  bf16*  hb    = (bf16*)nxt((size_t)CM * CHID * 2);           // hidden bf16 (reused as gated)
  bf16*  Wqb   = (bf16*)nxt((size_t)CNH * CD * CHID * 2);
  bf16*  Wkb   = (bf16*)nxt((size_t)CNKV * CD * CHID * 2);
  bf16*  Wvb   = (bf16*)nxt((size_t)CNKV * CD * CHID * 2);
  bf16*  Wgb   = (bf16*)nxt((size_t)CNH * CD * CHID * 2);
  bf16*  Wob   = (bf16*)nxt((size_t)CHID * CNH * CD * 2);
  float* qfb   = (float*)nxt((size_t)CM * CNH * CD * 4);      // reused as att
  float* kfb   = (float*)nxt((size_t)CM * CNKV * CD * 4);
  float* vfb   = (float*)nxt((size_t)CM * CNKV * CD * 4);
  float* gfb   = (float*)nxt((size_t)CM * CNH * CD * 4);
  bf16*  qTb   = (bf16*)nxt((size_t)CM * CNH * CD * 2);
  bf16*  kTb   = (bf16*)nxt((size_t)CM * CNKV * CD * 2);
  bf16*  vTb   = (bf16*)nxt((size_t)CM * CNKV * CD * 2);
  float* attf  = qfb;        // alias: q_f32 dead after norm_rope
  bf16*  gated = hb;         // alias: hidden_bf dead after projection GEMMs

  auto cvt = [&](const float* src, bf16* dst, int n) {
    int blocks = (n / 4 + 255) / 256;
    k_cvt_bf16<<<blocks, 256, 0, stream>>>(src, dst, n);
  };

  // 1) bf16 conversions (all L2-resident afterwards)
  cvt(hidden, hb,  CM * CHID);
  cvt(Wq,     Wqb, CNH * CD * CHID);
  cvt(Wk,     Wkb, CNKV * CD * CHID);
  cvt(Wv,     Wvb, CNKV * CD * CHID);
  cvt(Wg,     Wgb, CNH * CD * CHID);
  cvt(Wo,     Wob, CHID * CNH * CD);

  // 2) projection GEMMs: y = x @ W^T
  k_gemm_bf16<<<dim3((CNH * CD) / 128, CM / 128), 256, 0, stream>>>(
      hb, Wqb, qfb, CM, CNH * CD, CHID);
  k_gemm_bf16<<<dim3((CNKV * CD) / 128, CM / 128), 256, 0, stream>>>(
      hb, Wkb, kfb, CM, CNKV * CD, CHID);
  k_gemm_bf16<<<dim3((CNKV * CD) / 128, CM / 128), 256, 0, stream>>>(
      hb, Wvb, vfb, CM, CNKV * CD, CHID);
  k_gemm_bf16<<<dim3((CNH * CD) / 128, CM / 128), 256, 0, stream>>>(
      hb, Wgb, gfb, CM, CNH * CD, CHID);

  // 3) RMSNorm + RoPE + transpose to head-major bf16 (Q pre-scaled by 0.125)
  {
    int total = CM * (CNH + 2 * CNKV);
    k_norm_rope<<<(total + 255) / 256, 256, 0, stream>>>(
        qfb, kfb, vfb, cosb, sinb, qg, kg, qTb, kTb, vTb);
  }

  // 4) flash attention
  k_flash<<<dim3(CS / 128, CB * CNH), 256, 0, stream>>>(qTb, kTb, vTb, attf);

  // 5) sigmoid gating -> bf16
  {
    int n = CM * CNH * CD;
    k_gate<<<(n + 255) / 256, 256, 0, stream>>>(attf, gfb, gated, n);
  }

  // 6) output projection -> fp32 d_out
  k_gemm_bf16<<<dim3(CHID / 128, CM / 128), 256, 0, stream>>>(
      gated, Wob, out, CM, CHID, CNH * CD);
}
